// GMNConv_35862976922239
// MI455X (gfx1250) — matrix operand order repaired and checked by reading, not compiled, not run on hardware
//
#include <hip/hip_runtime.h>
#include <hip/hip_bf16.h>

#define NN 8192
#define EE 131072
#define BB 64
#define DD 128
#define MH 256
#define NEG_BIG -1e30f

typedef __bf16 bf16_t;
typedef __attribute__((ext_vector_type(16))) __bf16 v16bf;
typedef __attribute__((ext_vector_type(8)))  __bf16 v8bf;
typedef __attribute__((ext_vector_type(8)))  float  v8f;
typedef __attribute__((ext_vector_type(4)))  float  v4f;

__device__ __forceinline__ bf16_t f2bf(float f) {
  union { float f; unsigned u; } v; v.f = f;
  unsigned r = v.u + 0x7FFFu + ((v.u >> 16) & 1u);
  unsigned short h = (unsigned short)(r >> 16);
  bf16_t o;
  __builtin_memcpy(&o, &h, 2);
  return o;
}

// A-matrix (16-bit, 16x32): per lane, one contiguous 16-element run:
//   K = 2*koffA + e     (e=0..15)  koffA = (lane>=16) ? 8 : 0
// B-matrix (16-bit, 32x16): per lane, one contiguous 16-element run:
//   K = koffB + e       (e=0..15)  koffB = (lane>=16) ? 16 : 0
__device__ __forceinline__ v16bf make_a(const bf16_t* base, int koffA) {
  v8bf lo = *(const v8bf*)(base + 2 * koffA);
  v8bf hi = *(const v8bf*)(base + 2 * koffA + 8);
  v16bf a;
#pragma unroll
  for (int e = 0; e < 8; ++e) { a[e] = lo[e]; a[8 + e] = hi[e]; }
  return a;
}

// reuse_a: CDNA5 operand-cache hint for consecutive WMMAs sharing the A block
#define WMMA_BF16(a, b, c) \
  __builtin_amdgcn_wmma_f32_16x16x32_bf16(false, (a), false, (b), (short)0, (c), false, false)
#define WMMA_BF16_RA(a, b, c) \
  __builtin_amdgcn_wmma_f32_16x16x32_bf16(false, (a), false, (b), (short)0, (c), true, false)

// ---------------------------------------------------------------- utilities
__global__ void f2bf_kernel(const float* __restrict__ src, bf16_t* __restrict__ dst, int n) {
  int i = blockIdx.x * blockDim.x + threadIdx.x;
  if (i < n) dst[i] = f2bf(src[i]);
}

// transpose [n x d] f32 -> [d x n] bf16 (write-coalesced)
__global__ void xT_kernel(const float* __restrict__ x, bf16_t* __restrict__ xT, int n, int d) {
  int i = blockIdx.x * blockDim.x + threadIdx.x;
  if (i >= n * d) return;
  int c = i / n, r = i % n;
  xT[i] = f2bf(x[r * d + c]);
}

// swizzle W[K][N] row-major f32 into B-fragment blocks:
// out[((kt*(N/16)+nt)*32 + lane)*16 + e] = W[kt*32 + koffB + e][nt*16 + (lane&15)]
__global__ void wswz_kernel(const float* __restrict__ W, bf16_t* __restrict__ out, int K, int N) {
  int i = blockIdx.x * blockDim.x + threadIdx.x;
  if (i >= K * N) return;
  int e = i & 15;
  int lane = (i >> 4) & 31;
  int blk = i >> 9;
  int ntiles = N >> 4;
  int kt = blk / ntiles, nt = blk - kt * ntiles;
  int k = kt * 32 + e + ((lane >= 16) ? 16 : 0);
  int n = nt * 16 + (lane & 15);
  out[i] = f2bf(W[k * N + n]);
}

__device__ __forceinline__ int lower_bound_i(const int* a, int n, int key) {
  int lo = 0, hi = n;
  while (lo < hi) { int mid = (lo + hi) >> 1; if (a[mid] < key) lo = mid + 1; else hi = mid; }
  return lo;
}

__global__ void seg_kernel(const int* __restrict__ batch, int n, int* __restrict__ seg) {
  int b = threadIdx.x;
  if (b >= BB) return;
  seg[b * 2 + 0] = lower_bound_i(batch, n, b);
  seg[b * 2 + 1] = lower_bound_i(batch, n, b + 1);
}

// --------------------------------------------------- fused edge message MLP
__global__ __launch_bounds__(128) void msg_mlp_kernel(
    const float* __restrict__ x, const int* __restrict__ ei,
    const bf16_t* __restrict__ W1f, const float* __restrict__ b1,
    const bf16_t* __restrict__ W2f, const float* __restrict__ b2,
    float* __restrict__ messages) {
  __shared__ bf16_t a_lds[4][16 * 256];   // per-wave slice, reused for h
  const int wave = threadIdx.x >> 5;
  const int lane = threadIdx.x & 31;
  const int tile = blockIdx.x * 4 + wave;
  const int e0 = tile * 16;
  const int Ma = lane & 15;
  const int koffA = (lane >= 16) ? 8 : 0;

  const int myEdge = e0 + Ma;
  const int myTgt = ei[myEdge];
  const int mySrc = ei[EE + myEdge];

  // stage edge features [x_src | x_tgt] (16 x 256) as bf16, 8-wide chunks
  for (int ch = lane; ch < 512; ch += 32) {
    int r = ch >> 5;
    int c8 = (ch & 31) << 3;
    int s = __shfl(mySrc, r, 32);
    int t = __shfl(myTgt, r, 32);
    const float* sp = (c8 < DD) ? (x + (size_t)s * DD + c8)
                                : (x + (size_t)t * DD + (c8 - DD));
    v4f f0 = *(const v4f*)sp;
    v4f f1 = *(const v4f*)(sp + 4);
    v8bf o;
#pragma unroll
    for (int e = 0; e < 4; ++e) { o[e] = f2bf(f0[e]); o[4 + e] = f2bf(f1[e]); }
    *(v8bf*)&a_lds[wave][r * 256 + c8] = o;
  }
  __syncthreads();

  const v16bf* W1v = (const v16bf*)W1f;
  const v16bf* W2v = (const v16bf*)W2f;
  const v8f vz = {};
  v8f acc[16];
#pragma unroll
  for (int n = 0; n < 16; ++n) acc[n] = vz;
  {
    v16bf bm = W1v[lane];                        // (k=0, n=0)
    for (int k = 0; k < 8; ++k) {
      v16bf a = make_a(&a_lds[wave][Ma * 256 + k * 32], koffA);
#pragma unroll
      for (int n = 0; n < 16; ++n) {
        int nxt = k * 16 + n + 1;
        v16bf bnxt = W1v[((nxt < 128) ? nxt : 127) * 32 + lane];   // prefetch
        acc[n] = (n == 0) ? WMMA_BF16(a, bm, acc[n])
                          : WMMA_BF16_RA(a, bm, acc[n]);
        bm = bnxt;
      }
    }
  }
  __syncthreads();
#pragma unroll
  for (int n = 0; n < 16; ++n) {
#pragma unroll
    for (int r = 0; r < 8; ++r) {
      int row = r + ((lane >= 16) ? 8 : 0);
      int col = n * 16 + Ma;
      float v = acc[n][r] + b1[col];
      a_lds[wave][row * 256 + col] = f2bf(v > 0.f ? v : 0.f);
    }
  }
  __syncthreads();

  v8f acc2[16];
#pragma unroll
  for (int n = 0; n < 16; ++n) acc2[n] = vz;
  {
    v16bf bm = W2v[lane];
    for (int k = 0; k < 8; ++k) {
      v16bf a = make_a(&a_lds[wave][Ma * 256 + k * 32], koffA);
#pragma unroll
      for (int n = 0; n < 16; ++n) {
        int nxt = k * 16 + n + 1;
        v16bf bnxt = W2v[((nxt < 128) ? nxt : 127) * 32 + lane];
        acc2[n] = (n == 0) ? WMMA_BF16(a, bm, acc2[n])
                           : WMMA_BF16_RA(a, bm, acc2[n]);
        bm = bnxt;
      }
    }
  }
#pragma unroll
  for (int n = 0; n < 16; ++n) {
#pragma unroll
    for (int r = 0; r < 8; ++r) {
      int rM = r + ((lane >= 16) ? 8 : 0);
      int col = n * 16 + Ma;
      int trg = __shfl(myTgt, rM, 32);
      atomicAdd(&messages[(size_t)trg * MH + col], acc2[n][r] + b2[col]);
    }
  }
}

// ----------------------------------------- masked cross-attention (flash)
__global__ __launch_bounds__(32) void cross_attn_kernel(
    const float* __restrict__ xa, const bf16_t* __restrict__ xa_bf,
    const bf16_t* __restrict__ xb_bf, const bf16_t* __restrict__ xbT,
    const int* __restrict__ batcha, const int* __restrict__ batchb,
    const int* __restrict__ segb,
    float* __restrict__ att, int Nb) {
  __shared__ bf16_t p_lds[16 * 32];
  const int lane = threadIdx.x;
  const int i0 = blockIdx.x * 16;
  const int Ma = lane & 15;
  const int koffA = (lane >= 16) ? 8 : 0;
  const int koffB = (lane >= 16) ? 16 : 0;
  const int row = i0 + Ma;
  const int brow = batcha[row];

  const int bLo = __shfl(brow, 0, 32);
  const int bHi = __shfl(brow, 15, 32);
  int c0 = segb[bLo * 2 + 0] & ~31;
  int c1 = segb[bHi * 2 + 1];

  v16bf afr[4];
#pragma unroll
  for (int k = 0; k < 4; ++k)
    afr[k] = make_a(xa_bf + (size_t)row * DD + k * 32, koffA);

  const v8f vz = {};
  float m_r[8], l_r[8];
  v8f o[8];
#pragma unroll
  for (int r = 0; r < 8; ++r) { m_r[r] = NEG_BIG; l_r[r] = 0.f; o[r] = vz; }

  for (int j = c0; j < c1; j += 32) {
    v8f s[2];
#pragma unroll
    for (int t = 0; t < 2; ++t) {
      s[t] = vz;
      int col = j + t * 16 + Ma;
      int colc = (col < Nb) ? col : (Nb - 1);      // mask fixes OOB columns
      const bf16_t* cb = xb_bf + (size_t)colc * DD + koffB;
      v16bf bm = *(const v16bf*)cb;
#pragma unroll
      for (int k = 0; k < 4; ++k) {
        v16bf bnxt;
        if (k < 3) bnxt = *(const v16bf*)(cb + (k + 1) * 32);   // prefetch
        s[t] = WMMA_BF16(afr[k], bm, s[t]);
        if (k < 3) bm = bnxt;
      }
    }
    // mask + tile row-max
    float tmax[8];
#pragma unroll
    for (int r = 0; r < 8; ++r) tmax[r] = NEG_BIG;
#pragma unroll
    for (int t = 0; t < 2; ++t) {
      int col = j + t * 16 + Ma;
      int bcol = (col < Nb) ? batchb[col] : -1;
#pragma unroll
      for (int r = 0; r < 8; ++r) {
        int rM = r + ((lane >= 16) ? 8 : 0);
        int br = __shfl(brow, rM, 32);
        float v = (bcol == br) ? s[t][r] : NEG_BIG;
        s[t][r] = v;
        tmax[r] = fmaxf(tmax[r], v);
      }
    }
#pragma unroll
    for (int r = 0; r < 8; ++r)
#pragma unroll
      for (int msk = 1; msk <= 8; msk <<= 1)
        tmax[r] = fmaxf(tmax[r], __shfl_xor(tmax[r], msk, 32));
#pragma unroll
    for (int r = 0; r < 8; ++r) {
      float mn = fmaxf(m_r[r], tmax[r]);
      float sc = __expf(m_r[r] - mn);
      m_r[r] = mn;
      l_r[r] *= sc;
#pragma unroll
      for (int n = 0; n < 8; ++n) o[n][r] *= sc;
    }
#pragma unroll
    for (int t = 0; t < 2; ++t) {
#pragma unroll
      for (int r = 0; r < 8; ++r) {
        int rM = r + ((lane >= 16) ? 8 : 0);
        float p = __expf(s[t][r] - m_r[r]);
        float ps = p;
#pragma unroll
        for (int msk = 1; msk <= 8; msk <<= 1)
          ps += __shfl_xor(ps, msk, 32);
        l_r[r] += ps;
        p_lds[rM * 32 + t * 16 + Ma] = f2bf(p);
      }
    }
    __syncthreads();
    // o += P(16x32) @ xb[j:j+32, :128]  (B-frags from transposed xb)
    v16bf pa = make_a(&p_lds[Ma * 32], koffA);
    v16bf bm = *(const v16bf*)(xbT + (size_t)Ma * NN + j + koffB);
#pragma unroll
    for (int n = 0; n < 8; ++n) {
      v16bf bnxt;
      if (n < 7) bnxt = *(const v16bf*)(xbT + (size_t)((n + 1) * 16 + Ma) * NN + j + koffB);
      o[n] = (n == 0) ? WMMA_BF16(pa, bm, o[n]) : WMMA_BF16_RA(pa, bm, o[n]);
      if (n < 7) bm = bnxt;
    }
    __syncthreads();
  }
#pragma unroll
  for (int n = 0; n < 8; ++n) {
#pragma unroll
    for (int r = 0; r < 8; ++r) {
      int rM = r + ((lane >= 16) ? 8 : 0);
      int col = n * 16 + Ma;
      int rw = i0 + rM;
      float denom = (l_r[r] > 0.f) ? l_r[r] : 1.f;
      att[(size_t)rw * DD + col] = xa[(size_t)rw * DD + col] - o[n][r] / denom;
    }
  }
}

// --------------------------------------------------- fused node update MLP
__global__ __launch_bounds__(128) void update_kernel(
    const float* __restrict__ x, const float* __restrict__ messages,
    const float* __restrict__ att,
    const bf16_t* __restrict__ W1f, const float* __restrict__ b1,
    const bf16_t* __restrict__ W2f, const float* __restrict__ b2,
    float* __restrict__ out) {
  __shared__ bf16_t u_lds[4][16 * 512];   // reused for h (first 16*256)
  const int wave = threadIdx.x >> 5;
  const int lane = threadIdx.x & 31;
  const int tile = blockIdx.x * 4 + wave;
  const int n0 = tile * 16;
  const int Ma = lane & 15;
  const int koffA = (lane >= 16) ? 8 : 0;

  // stage u = [messages | att | x] (16 x 512) as bf16, 8-wide chunks
  for (int ch = lane; ch < 1024; ch += 32) {
    int r = ch >> 6;
    int c8 = (ch & 63) << 3;
    int node = n0 + r;
    const float* sp;
    if (c8 < 256)      sp = messages + (size_t)node * MH + c8;
    else if (c8 < 384) sp = att + (size_t)node * DD + (c8 - 256);
    else               sp = x + (size_t)node * DD + (c8 - 384);
    v4f f0 = *(const v4f*)sp;
    v4f f1 = *(const v4f*)(sp + 4);
    v8bf o;
#pragma unroll
    for (int e = 0; e < 4; ++e) { o[e] = f2bf(f0[e]); o[4 + e] = f2bf(f1[e]); }
    *(v8bf*)&u_lds[wave][r * 512 + c8] = o;
  }
  __syncthreads();

  const v16bf* W1v = (const v16bf*)W1f;
  const v16bf* W2v = (const v16bf*)W2f;
  const v8f vz = {};
  v8f acc[16];
#pragma unroll
  for (int n = 0; n < 16; ++n) acc[n] = vz;
  {
    v16bf bm = W1v[lane];
    for (int k = 0; k < 16; ++k) {   // K = 512
      v16bf a = make_a(&u_lds[wave][Ma * 512 + k * 32], koffA);
#pragma unroll
      for (int n = 0; n < 16; ++n) {
        int nxt = k * 16 + n + 1;
        v16bf bnxt = W1v[((nxt < 256) ? nxt : 255) * 32 + lane];
        acc[n] = (n == 0) ? WMMA_BF16(a, bm, acc[n])
                          : WMMA_BF16_RA(a, bm, acc[n]);
        bm = bnxt;
      }
    }
  }
  __syncthreads();
#pragma unroll
  for (int n = 0; n < 16; ++n) {
#pragma unroll
    for (int r = 0; r < 8; ++r) {
      int row = r + ((lane >= 16) ? 8 : 0);
      int col = n * 16 + Ma;
      float v = acc[n][r] + b1[col];
      u_lds[wave][row * 256 + col] = f2bf(v > 0.f ? v : 0.f);
    }
  }
  __syncthreads();

  v8f acc2[8];
#pragma unroll
  for (int n = 0; n < 8; ++n) acc2[n] = vz;
  {
    v16bf bm = W2v[lane];
    for (int k = 0; k < 8; ++k) {    // K = 256, N = 128
      v16bf a = make_a(&u_lds[wave][Ma * 256 + k * 32], koffA);
#pragma unroll
      for (int n = 0; n < 8; ++n) {
        int nxt = k * 8 + n + 1;
        v16bf bnxt = W2v[((nxt < 64) ? nxt : 63) * 32 + lane];
        acc2[n] = (n == 0) ? WMMA_BF16(a, bm, acc2[n])
                           : WMMA_BF16_RA(a, bm, acc2[n]);
        bm = bnxt;
      }
    }
  }
#pragma unroll
  for (int n = 0; n < 8; ++n) {
#pragma unroll
    for (int r = 0; r < 8; ++r) {
      int rM = r + ((lane >= 16) ? 8 : 0);
      int col = n * 16 + Ma;
      int node = n0 + rM;
      out[(size_t)node * DD + col] = x[(size_t)node * DD + col] + acc2[n][r] + b2[col];
    }
  }
}

// ----------------------------------------------------------------- launch
extern "C" void kernel_launch(void* const* d_in, const int* in_sizes, int n_in,
                              void* d_out, int out_size, void* d_ws, size_t ws_size,
                              hipStream_t stream) {
  (void)in_sizes; (void)n_in; (void)out_size; (void)ws_size;
  const float* x1     = (const float*)d_in[0];
  const int*   ei1    = (const int*)d_in[1];
  const int*   batch1 = (const int*)d_in[2];
  const float* x2     = (const float*)d_in[3];
  const int*   ei2    = (const int*)d_in[4];
  const int*   batch2 = (const int*)d_in[5];
  const float* msgW1  = (const float*)d_in[6];
  const float* msgb1  = (const float*)d_in[7];
  const float* msgW2  = (const float*)d_in[8];
  const float* msgb2  = (const float*)d_in[9];
  const float* updW1  = (const float*)d_in[10];
  const float* updb1  = (const float*)d_in[11];
  const float* updW2  = (const float*)d_in[12];
  const float* updb2  = (const float*)d_in[13];

  char* ws = (char*)d_ws;
  auto carve = [&](size_t bytes) {
    char* p = ws;
    ws += (bytes + 255) & ~(size_t)255;
    return (void*)p;
  };
  bf16_t* x1b  = (bf16_t*)carve((size_t)NN * DD * 2);
  bf16_t* x2b  = (bf16_t*)carve((size_t)NN * DD * 2);
  bf16_t* x1T  = (bf16_t*)carve((size_t)NN * DD * 2 + 256);  // pad: P@V tail reads
  bf16_t* x2T  = (bf16_t*)carve((size_t)NN * DD * 2 + 256);
  bf16_t* mW1f = (bf16_t*)carve((size_t)256 * 256 * 2);
  bf16_t* mW2f = (bf16_t*)carve((size_t)256 * 256 * 2);
  bf16_t* uW1f = (bf16_t*)carve((size_t)512 * 256 * 2);
  bf16_t* uW2f = (bf16_t*)carve((size_t)256 * 128 * 2);
  float* msgs1 = (float*)carve((size_t)NN * MH * 4);
  float* msgs2 = (float*)carve((size_t)NN * MH * 4);
  float* att1  = (float*)carve((size_t)NN * DD * 4);
  float* att2  = (float*)carve((size_t)NN * DD * 4);
  int* seg1    = (int*)carve((size_t)BB * 2 * 4);
  int* seg2    = (int*)carve((size_t)BB * 2 * 4);

  f2bf_kernel<<<(NN * DD + 255) / 256, 256, 0, stream>>>(x1, x1b, NN * DD);
  f2bf_kernel<<<(NN * DD + 255) / 256, 256, 0, stream>>>(x2, x2b, NN * DD);
  xT_kernel<<<(NN * DD + 255) / 256, 256, 0, stream>>>(x1, x1T, NN, DD);
  xT_kernel<<<(NN * DD + 255) / 256, 256, 0, stream>>>(x2, x2T, NN, DD);
  wswz_kernel<<<(256 * 256 + 255) / 256, 256, 0, stream>>>(msgW1, mW1f, 256, 256);
  wswz_kernel<<<(256 * 256 + 255) / 256, 256, 0, stream>>>(msgW2, mW2f, 256, 256);
  wswz_kernel<<<(512 * 256 + 255) / 256, 256, 0, stream>>>(updW1, uW1f, 512, 256);
  wswz_kernel<<<(256 * 128 + 255) / 256, 256, 0, stream>>>(updW2, uW2f, 256, 128);

  seg_kernel<<<1, 64, 0, stream>>>(batch1, NN, seg1);
  seg_kernel<<<1, 64, 0, stream>>>(batch2, NN, seg2);

  hipMemsetAsync(msgs1, 0, (size_t)NN * MH * 4, stream);
  hipMemsetAsync(msgs2, 0, (size_t)NN * MH * 4, stream);

  msg_mlp_kernel<<<EE / 64, 128, 0, stream>>>(x1, ei1, mW1f, msgb1, mW2f, msgb2, msgs1);
  msg_mlp_kernel<<<EE / 64, 128, 0, stream>>>(x2, ei2, mW1f, msgb1, mW2f, msgb2, msgs2);

  cross_attn_kernel<<<NN / 16, 32, 0, stream>>>(x1, x1b, x2b, x2T, batch1, batch2, seg2, att1, NN);
  cross_attn_kernel<<<NN / 16, 32, 0, stream>>>(x2, x2b, x1b, x1T, batch2, batch1, seg1, att2, NN);

  float* out1 = (float*)d_out;
  float* out2 = out1 + (size_t)NN * DD;
  update_kernel<<<NN / 64, 128, 0, stream>>>(x1, msgs1, att1, uW1f, updb1, uW2f, updb2, out1);
  update_kernel<<<NN / 64, 128, 0, stream>>>(x2, msgs2, att2, uW1f, updb1, uW2f, updb2, out2);
}
